// HeteroGNN_24404004176459
// MI455X (gfx1250) — matrix operand order repaired and checked by reading, not compiled, not run on hardware
//
#include <hip/hip_runtime.h>
#include <hip/hip_bf16.h>

typedef __attribute__((ext_vector_type(16))) __bf16 v16bf;
typedef __attribute__((ext_vector_type(8)))  __bf16 v8bf;
typedef __attribute__((ext_vector_type(8)))  float  v8f;

#define SLD 136   // padded LDS row stride (elements) to avoid bank conflicts
#define ROWS 64   // output rows per block (4 x 16-row WMMA sub-tiles)

// ---------------------------------------------------------------------------
// degree counts: cnt[dst[e]] += 1.0f
__global__ void count_kernel(const int* __restrict__ dst, float* __restrict__ cnt, int E) {
    int e = blockIdx.x * blockDim.x + threadIdx.x;
    if (e < E) atomicAdd(&cnt[dst[e]], 1.0f);
}

// gather + scatter-add of 128-float rows: agg[dst[e]] += x[src[e]]
// one wave32 per edge, 4 floats per lane (float4 coalesced load)
__global__ __launch_bounds__(256)
void scatter_add_rows(const float* __restrict__ x, const int* __restrict__ src,
                      const int* __restrict__ dst, float* __restrict__ agg, int E) {
    int gw   = (blockIdx.x * blockDim.x + threadIdx.x) >> 5;
    int lane = threadIdx.x & 31;
    if (gw >= E) return;
    int s = src[gw];
    int d = dst[gw];
    const float4 v = *(const float4*)(x + (size_t)s * 128 + lane * 4);
    float* o = agg + (size_t)d * 128 + lane * 4;
    atomicAdd(o + 0, v.x);
    atomicAdd(o + 1, v.y);
    atomicAdd(o + 2, v.z);
    atomicAdd(o + 3, v.w);
}

// elementwise o = a + b  (used to pre-sum Wr pairs and bias pairs)
__global__ void add_vec(const float* __restrict__ a, const float* __restrict__ b,
                        float* __restrict__ o, int n) {
    int i = blockIdx.x * blockDim.x + threadIdx.x;
    if (i < n) o[i] = a[i] + b[i];
}

// ---------------------------------------------------------------------------
// out[N,128] (+)= relu?( A[N,K] (row-scaled by 1/max(cnt,1)) @ W[K,128] + bias )
// bf16 WMMA 16x16x32, f32 accumulate. Block = 256 thr = 8 waves.
// Block computes rows [64*bx, 64*bx+64); wave w computes cols [16w,16w+16)
// across all 4 row sub-tiles, reusing the B fragment per K-step.
__global__ __launch_bounds__(256)
void gemm_wmma_bf16(const float* __restrict__ A, const float* __restrict__ Wm,
                    const float* __restrict__ bias, const float* __restrict__ cnt,
                    float* __restrict__ out, int N, int K, int kshift,
                    int accum, int relu) {
    __shared__ __bf16 sW[128 * SLD];   // W transposed: sW[n*SLD + k] = W[k][n]
    __shared__ __bf16 sA[ROWS * SLD];  // sA[r*SLD + c] = A_tile[r][c] (pre-scaled)

    const int tid  = threadIdx.x;
    const int row0 = blockIdx.x * ROWS;

    // load W (coalesced over n), store transposed bf16
    for (int idx = tid; idx < K * 128; idx += 256) {
        int k = idx >> 7;          // idx / 128
        int n = idx & 127;
        sW[n * SLD + k] = (__bf16)Wm[idx];
    }
    // load A tile with optional mean-scaling (K is 64 or 128 -> shifts)
    const int kmask = K - 1;
    for (int idx = tid; idx < ROWS * K; idx += 256) {
        int r = idx >> kshift;
        int c = idx & kmask;
        int row = row0 + r;
        float v = 0.0f;
        if (row < N) {
            v = A[(size_t)row * K + c];
            if (cnt) v *= 1.0f / fmaxf(cnt[row], 1.0f);
        }
        sA[r * SLD + c] = (__bf16)v;
    }
    __syncthreads();

    const int lane = tid & 31;
    const int wv   = tid >> 5;
    const int hf   = lane >> 4;        // half of the wave
    const int m    = lane & 15;        // A row within 16-row sub-tile
    const int ncol = wv * 16 + (lane & 15);

    v8f acc0 = {}, acc1 = {}, acc2 = {}, acc3 = {};
    for (int kk = 0; kk < K; kk += 32) {
        // B fragment: b[0..15] = K kk+hf*16+{0..15} at column ncol (reused 4x)
        const __bf16* pb = sW + ncol * SLD + kk;
        v8bf blo = *(const v8bf*)(pb + hf * 16);
        v8bf bhi = *(const v8bf*)(pb + hf * 16 + 8);
        v16bf b;
#pragma unroll
        for (int q = 0; q < 8; ++q) { b[q] = blo[q]; b[8 + q] = bhi[q]; }

#pragma unroll
        for (int t = 0; t < 4; ++t) {
            // A fragment for sub-tile t: a[0..15]=K kk+hf*16+{0..15}
            const __bf16* pa = sA + (t * 16 + m) * SLD + kk;
            v8bf alo = *(const v8bf*)(pa + hf * 16);
            v8bf ahi = *(const v8bf*)(pa + hf * 16 + 8);
            v16bf a;
#pragma unroll
            for (int q = 0; q < 8; ++q) { a[q] = alo[q]; a[8 + q] = ahi[q]; }
            v8f c = (t == 0) ? acc0 : (t == 1) ? acc1 : (t == 2) ? acc2 : acc3;
            c = __builtin_amdgcn_wmma_f32_16x16x32_bf16(
                    false, a, false, b, (short)0, c, false, false);
            if      (t == 0) acc0 = c;
            else if (t == 1) acc1 = c;
            else if (t == 2) acc2 = c;
            else             acc3 = c;
        }
    }

    // epilogue: VGPR r of C/D holds row m = r + 8*hf (ISA 7.12.2)
#pragma unroll
    for (int t = 0; t < 4; ++t) {
        v8f c = (t == 0) ? acc0 : (t == 1) ? acc1 : (t == 2) ? acc2 : acc3;
#pragma unroll
        for (int r = 0; r < 8; ++r) {
            int row = row0 + t * 16 + r + hf * 8;
            if (row < N) {
                size_t o = (size_t)row * 128 + ncol;
                float v = c[r];
                if (bias)  v += bias[ncol];
                if (accum) v += out[o];
                if (relu)  v = fmaxf(v, 0.0f);
                out[o] = v;
            }
        }
    }
}

// ---------------------------------------------------------------------------
// out[i] = dot(X[i,:128], Wc) + bc ; one wave per row, shfl_xor reduction
__global__ __launch_bounds__(256)
void classify(const float* __restrict__ X, const float* __restrict__ Wc,
              const float* __restrict__ bc, float* __restrict__ out, int N) {
    int gw   = (blockIdx.x * blockDim.x + threadIdx.x) >> 5;
    int lane = threadIdx.x & 31;
    if (gw >= N) return;
    const float4 x = *(const float4*)(X  + (size_t)gw * 128 + lane * 4);
    const float4 w = *(const float4*)(Wc + lane * 4);
    float s = x.x * w.x + x.y * w.y + x.z * w.z + x.w * w.w;
    for (int off = 16; off; off >>= 1) s += __shfl_xor(s, off, 32);
    if (lane == 0) out[gw] = s + bc[0];
}

// ---------------------------------------------------------------------------
extern "C" void kernel_launch(void* const* d_in, const int* in_sizes, int n_in,
                              void* d_out, int out_size, void* d_ws, size_t ws_size,
                              hipStream_t stream) {
    const float* x_tx = (const float*)d_in[0];
    const float* W_in = (const float*)d_in[1];
    const float* b_in = (const float*)d_in[2];
    const float* Wl1  = (const float*)d_in[3];   // [4,128,128]
    const float* bl1  = (const float*)d_in[4];   // [4,128]
    const float* Wr1  = (const float*)d_in[5];
    const float* Wl2  = (const float*)d_in[6];
    const float* bl2  = (const float*)d_in[7];
    const float* Wr2  = (const float*)d_in[8];
    const float* Wc   = (const float*)d_in[9];
    const float* bc   = (const float*)d_in[10];
    const int* tc_src = (const int*)d_in[11];
    const int* tc_dst = (const int*)d_in[12];
    const int* te_src = (const int*)d_in[13];
    const int* te_dst = (const int*)d_in[14];

    const int E  = in_sizes[11];
    const int NT = 200000, NC = 80000, NE = 60000;
    const int H = 128, F = 64, HH = H * H;

    // ---- workspace layout ----
    char* base = (char*)d_ws;
    size_t off = 0;
    auto take = [&](size_t bytes) -> char* {
        char* p = base + off;
        off = (off + bytes + 255) & ~(size_t)255;
        return p;
    };
    float* B0      = (float*)take((size_t)NT * H * 4);  // h, later tx2
    float* tx1     = (float*)take((size_t)NT * H * 4);
    float* agg     = (float*)take((size_t)NT * H * 4);  // shared agg scratch
    float* card1   = (float*)take((size_t)NC * H * 4);
    float* email1  = (float*)take((size_t)NE * H * 4);
    float* cnts    = (float*)take((size_t)(NC + NE + NT + NT) * 4);
    float* cntCard = cnts;
    float* cntEml  = cnts + NC;
    float* cntTxC  = cnts + NC + NE;
    float* cntTxE  = cnts + NC + NE + NT;
    float* Wsum1   = (float*)take((size_t)HH * 4);
    float* bsum1   = (float*)take((size_t)H * 4);
    float* Wsum2   = (float*)take((size_t)HH * 4);
    float* bsum2   = (float*)take((size_t)H * 4);

    const int TPB = 256;
    const int egrid = (E + TPB - 1) / TPB;   // thread-per-edge grids
    const int wgrid = (E + 7) / 8;           // wave-per-edge grids
    const int gT = (NT + ROWS - 1) / ROWS;
    const int gC = (NC + ROWS - 1) / ROWS;
    const int gE = (NE + ROWS - 1) / ROWS;

    // 1) counts (graph structure fixed per launch)
    hipMemsetAsync(cnts, 0, (size_t)(NC + NE + NT + NT) * 4, stream);
    count_kernel<<<egrid, TPB, 0, stream>>>(tc_dst, cntCard, E);
    count_kernel<<<egrid, TPB, 0, stream>>>(te_dst, cntEml, E);
    count_kernel<<<egrid, TPB, 0, stream>>>(tc_src, cntTxC, E);
    count_kernel<<<egrid, TPB, 0, stream>>>(te_src, cntTxE, E);

    // 2) pre-sum weight/bias pairs (tx receives two edge types over the same x)
    add_vec<<<(HH + TPB - 1) / TPB, TPB, 0, stream>>>(Wr1 + 1 * HH, Wr1 + 3 * HH, Wsum1, HH);
    add_vec<<<1, TPB, 0, stream>>>(bl1 + 1 * H, bl1 + 3 * H, bsum1, H);
    add_vec<<<(HH + TPB - 1) / TPB, TPB, 0, stream>>>(Wr2 + 1 * HH, Wr2 + 3 * HH, Wsum2, HH);
    add_vec<<<1, TPB, 0, stream>>>(bl2 + 1 * H, bl2 + 3 * H, bsum2, H);

    // 3) h = x_tx @ W_in + b_in
    gemm_wmma_bf16<<<gT, TPB, 0, stream>>>(
        x_tx, W_in, b_in, nullptr, B0, NT, F, 6, 0, 0);

    // 4) layer-1 tx: neighbor feats are zero -> tx1 = relu(h@(Wr1_1+Wr1_3) + b1+b3)
    gemm_wmma_bf16<<<gT, TPB, 0, stream>>>(
        B0, Wsum1, bsum1, nullptr, tx1, NT, H, 7, 0, 1);

    // 5) card1 = relu(mean_{tc}(h[tc_src] -> tc_dst) @ Wl1_0 + bl1_0)
    hipMemsetAsync(agg, 0, (size_t)NC * H * 4, stream);
    scatter_add_rows<<<wgrid, TPB, 0, stream>>>(B0, tc_src, tc_dst, agg, E);
    gemm_wmma_bf16<<<gC, TPB, 0, stream>>>(
        agg, Wl1 + 0 * HH, bl1 + 0 * H, cntCard, card1, NC, H, 7, 0, 1);

    // 6) email1 = relu(mean_{te}(h[te_src] -> te_dst) @ Wl1_2 + bl1_2)
    hipMemsetAsync(agg, 0, (size_t)NE * H * 4, stream);
    scatter_add_rows<<<wgrid, TPB, 0, stream>>>(B0, te_src, te_dst, agg, E);
    gemm_wmma_bf16<<<gE, TPB, 0, stream>>>(
        agg, Wl1 + 2 * HH, bl1 + 2 * H, cntEml, email1, NE, H, 7, 0, 1);

    // 7) layer-2 tx self term: tx2 = tx1@(Wr2_1+Wr2_3) + b1+b3   (B0 now dead -> tx2)
    float* tx2 = B0;
    gemm_wmma_bf16<<<gT, TPB, 0, stream>>>(
        tx1, Wsum2, bsum2, nullptr, tx2, NT, H, 7, 0, 0);

    // 8) tx2 += mean_{tc}(card1[tc_dst] -> tc_src) @ Wl2_1
    hipMemsetAsync(agg, 0, (size_t)NT * H * 4, stream);
    scatter_add_rows<<<wgrid, TPB, 0, stream>>>(card1, tc_dst, tc_src, agg, E);
    gemm_wmma_bf16<<<gT, TPB, 0, stream>>>(
        agg, Wl2 + 1 * HH, nullptr, cntTxC, tx2, NT, H, 7, 1, 0);

    // 9) tx2 += mean_{te}(email1[te_dst] -> te_src) @ Wl2_3 ; final relu
    hipMemsetAsync(agg, 0, (size_t)NT * H * 4, stream);
    scatter_add_rows<<<wgrid, TPB, 0, stream>>>(email1, te_dst, te_src, agg, E);
    gemm_wmma_bf16<<<gT, TPB, 0, stream>>>(
        agg, Wl2 + 3 * HH, nullptr, cntTxE, tx2, NT, H, 7, 1, 1);

    // 10) out = tx2 @ Wc + bc
    classify<<<(NT + 7) / 8, TPB, 0, stream>>>(tx2, Wc, bc, (float*)d_out, NT);
}
